// SentenceEncodingRNN2_41283225649717
// MI455X (gfx1250) — matrix operand order repaired and checked by reading, not compiled, not run on hardware
//
#include <hip/hip_runtime.h>
#include <stdint.h>

// ---------------------------------------------------------------------------
// CDNA5 (gfx1250) WMMA bf16 pipeline for a 2-layer BiLSTM + attention pooling.
// T=256, B=128, D=300, H=256.
//  - all GEMMs on v_wmma_f32_16x16x32_bf16
//  - recurrent xg slabs streamed with global_load_async_to_lds_b128 (ASYNCcnt)
// ---------------------------------------------------------------------------

#define TT 256
#define BB 128
#define DD 300
#define HH 256
#define MROWS (TT * BB)      // 32768
#define G4H (4 * HH)         // 1024
#define H2 (2 * HH)          // 512
#define KP0 320              // D padded to multiple of 32

typedef __bf16 bf16x16 __attribute__((ext_vector_type(16)));
typedef float f32x8 __attribute__((ext_vector_type(8)));

union Frag {
    bf16x16 v;
    uint4 q[2];
};

__device__ __forceinline__ unsigned short f2bf(float f) {
    unsigned u = __float_as_uint(f);
    unsigned r = (u + 0x7FFFu + ((u >> 16) & 1u)) >> 16;
    return (unsigned short)r;
}

__device__ __forceinline__ float bf2f(unsigned short u) {
    return __uint_as_float(((unsigned)u) << 16);
}

__device__ __forceinline__ f32x8 wmma_bf16(const Frag& a, const Frag& b, f32x8 c) {
    return __builtin_amdgcn_wmma_f32_16x16x32_bf16(
        false, a.v, false, b.v, (short)0, c, false, false);
}

__device__ __forceinline__ float sigf(float x) {
    return 1.0f / (1.0f + __expf(-x));
}

// async 16B global -> LDS copy (per-lane addresses), tracked by ASYNCcnt
__device__ __forceinline__ void async_cp16(unsigned ldsOff, const void* gptr) {
    asm volatile("global_load_async_to_lds_b128 %0, %1, off"
                 :: "v"(ldsOff), "v"(gptr)
                 : "memory");
}
__device__ __forceinline__ void wait_async0() {
    asm volatile("s_wait_asynccnt 0x0" ::: "memory");
}

// ---------------------------------------------------------------------------
// fp32 -> bf16 convert with K padding (zero fill)
// ---------------------------------------------------------------------------
__global__ void cvt_pad_bf16(const float* __restrict__ src,
                             unsigned short* __restrict__ dst,
                             int K, int Kpad, long total) {
    long idx = (long)blockIdx.x * blockDim.x + threadIdx.x;
    if (idx >= total) return;
    long n = idx / Kpad;
    int k = (int)(idx - n * Kpad);
    dst[idx] = (k < K) ? f2bf(src[n * K + k]) : (unsigned short)0;
}

__global__ void zerof(float* __restrict__ p, int n) {
    int i = blockIdx.x * blockDim.x + threadIdx.x;
    if (i < n) p[i] = 0.0f;
}

// ---------------------------------------------------------------------------
// WMMA GEMM: acc[M,N] = A[M,K](bf16) @ W[N,K](bf16)^T + bias0 + bias1
//   EPI==0: store fp32 C
//   EPI==2: store bf16 Cbf
//   EPI==1: ann = tanh(acc+bias); scores[row] += sum_col ann*ctx[col] (atomic)
// block 256 thr = 8 waves; wave -> 16x64 strip; block tile 32x256.
// grid: (M/32, N/256)
// ---------------------------------------------------------------------------
template <int EPI>
__global__ __launch_bounds__(256) void gemm_bf16(
    const unsigned short* __restrict__ A, int ldA,
    const unsigned short* __restrict__ W, int ldW,
    const float* __restrict__ bias0, const float* __restrict__ bias1,
    float* __restrict__ C, unsigned short* __restrict__ Cbf, int ldC, int K,
    const float* __restrict__ ctx, float* __restrict__ scores) {
    const int wave = threadIdx.x >> 5;
    const int lane = threadIdx.x & 31;
    const int half = lane >> 4;
    const int r = lane & 15;

    const int mT = blockIdx.x * 2 + (wave & 1);         // 16-row tile index
    const int nColBase = blockIdx.y * 256 + (wave >> 1) * 64;

    f32x8 acc[4];
#pragma unroll
    for (int i = 0; i < 4; ++i)
#pragma unroll
        for (int j = 0; j < 8; ++j) acc[i][j] = 0.0f;

    const unsigned short* pA = A + (size_t)(mT * 16 + r) * ldA + half * 8;
    const unsigned short* pB0 = W + (size_t)(nColBase + r) * ldW + half * 8;

    for (int k0 = 0; k0 < K; k0 += 32) {
        Frag a;
        a.q[0] = *(const uint4*)(pA + k0);
        a.q[1] = *(const uint4*)(pA + k0 + 16);
        if (k0 + 32 < K) __builtin_prefetch(pA + k0 + 32, 0, 1);
#pragma unroll
        for (int i = 0; i < 4; ++i) {
            const unsigned short* pB = pB0 + (size_t)(i * 16) * ldW + k0;
            Frag b;
            b.q[0] = *(const uint4*)(pB);
            b.q[1] = *(const uint4*)(pB + 16);
            acc[i] = wmma_bf16(a, b, acc[i]);
        }
    }

    if (EPI == 0 || EPI == 2) {
#pragma unroll
        for (int i = 0; i < 4; ++i) {
            const int col = nColBase + i * 16 + r;
            float bb = (bias0 ? bias0[col] : 0.0f) + (bias1 ? bias1[col] : 0.0f);
#pragma unroll
            for (int r8 = 0; r8 < 8; ++r8) {
                int row = mT * 16 + half * 8 + r8;
                if (EPI == 0)
                    C[(size_t)row * ldC + col] = acc[i][r8] + bb;
                else
                    Cbf[(size_t)row * ldC + col] = f2bf(acc[i][r8] + bb);
            }
        }
    } else {
        float sAcc[8];
#pragma unroll
        for (int r8 = 0; r8 < 8; ++r8) sAcc[r8] = 0.0f;
#pragma unroll
        for (int i = 0; i < 4; ++i) {
            const int col = nColBase + i * 16 + r;
            float bb = bias0 ? bias0[col] : 0.0f;
            float cx = ctx[col];
#pragma unroll
            for (int r8 = 0; r8 < 8; ++r8)
                sAcc[r8] += tanhf(acc[i][r8] + bb) * cx;
        }
#pragma unroll
        for (int r8 = 0; r8 < 8; ++r8) {
            float v = sAcc[r8];
#pragma unroll
            for (int m = 1; m < 16; m <<= 1) v += __shfl_xor(v, m, 32);
            if (r == 0) {
                int row = mT * 16 + half * 8 + r8;
                atomicAdd(scores + row, v);
            }
        }
    }
}

// ---------------------------------------------------------------------------
// One bidirectional LSTM layer (packed-sequence semantics via time involution).
// grid = 8 blocks: blockIdx.x = dir*4 + batchBlock (32 batch rows each).
// 1024 threads = 32 waves; wave owns one 16x16 h-tile (mT in {0,1}, nT in 0..15)
// and its 4 gate tiles.  c stays in VGPRs, h lives in LDS (bf16, stride 272).
// Per step: issue async copy of the 32x1024 bf16 xg slab into LDS, run the
// recurrent WMMAs on h(t-1) while the copy is in flight, then wait ASYNCcnt,
// barrier, and finish the gates from LDS.
// ---------------------------------------------------------------------------
#define HLD 272

__global__ __launch_bounds__(1024) void lstm_bidir(
    const unsigned short* __restrict__ whhF,
    const unsigned short* __restrict__ whhR,
    const unsigned short* __restrict__ xgF,   // [T*B, 1024] bf16 (bias folded)
    const unsigned short* __restrict__ xgR,
    const int* __restrict__ lengths,
    unsigned short* __restrict__ outBf,       // [T*B, 512] bf16
    float* __restrict__ outF32) {             // optional [T*B, 512] fp32
    const int dir = blockIdx.x >> 2;
    const int bBase = (blockIdx.x & 3) * 32;
    const unsigned short* whh = dir ? whhR : whhF;
    const unsigned short* xg = dir ? xgR : xgF;

    __shared__ unsigned short hP[32 * HLD];
    __shared__ unsigned short xgS[32 * G4H];   // 64KB staging slab
    __shared__ int lenS[32];

    for (int i = threadIdx.x; i < 32 * HLD; i += 1024) hP[i] = 0;
    if (threadIdx.x < 32) {
        int l = lengths[bBase + threadIdx.x];
        lenS[threadIdx.x] = l < 1 ? 1 : l;
    }
    __syncthreads();

    const int tid = threadIdx.x;
    const int wave = tid >> 5;
    const int lane = tid & 31;
    const int half = lane >> 4;
    const int r = lane & 15;
    const int mT = wave >> 4;     // 0..1  (16-row local tile)
    const int nT = wave & 15;     // 0..15 (h-column tile)
    const int hcol = nT * 16 + r;

    f32x8 cReg;
#pragma unroll
    for (int j = 0; j < 8; ++j) cReg[j] = 0.0f;

    const unsigned short* pA = hP + (mT * 16 + r) * HLD + half * 8;
    const unsigned xgSbase = (unsigned)(uintptr_t)(&xgS[0]);

    for (int t = 0; t < TT; ++t) {
        // Launder a zero offset each iteration: defeats LICM of the
        // (t-invariant) W_hh fragment loads (no 256-VGPR hoist + spill) while
        // preserving the global address space (global_load_b128, not flat).
        unsigned zoff = 0;
        asm volatile("" : "+v"(zoff));
        const unsigned short* whh_t = whh + zoff;

        // ---- issue async copy of xg slab for this step (overlaps WMMAs) ----
        // 32 rows x 2048B = 4096 16B-chunks; 1024 threads x 4 chunks.
#pragma unroll
        for (int j = 0; j < 4; ++j) {
            int chunk = j * 1024 + tid;
            int lrow = chunk >> 7;          // 128 chunks per row
            int c16 = chunk & 127;
            int len = lenS[lrow];
            int tw = dir ? ((t < len) ? (len - 1 - t) : t) : t;
            const unsigned short* g =
                xg + ((size_t)tw * BB + bBase + lrow) * G4H + c16 * 8;
            async_cp16(xgSbase + (unsigned)(lrow * G4H + c16 * 8) * 2, g);
        }

        // ---- recurrent WMMAs on h(t-1) ----
        f32x8 acc[4];
#pragma unroll
        for (int g = 0; g < 4; ++g)
#pragma unroll
            for (int j = 0; j < 8; ++j) acc[g][j] = 0.0f;

        for (int k0 = 0; k0 < HH; k0 += 32) {
            Frag a;
            a.q[0] = *(const uint4*)(pA + k0);
            a.q[1] = *(const uint4*)(pA + k0 + 16);
#pragma unroll
            for (int g = 0; g < 4; ++g) {
                const unsigned short* pB =
                    whh_t + (size_t)(g * HH + nT * 16 + r) * HH + k0 + half * 8;
                Frag b;
                b.q[0] = *(const uint4*)(pB);
                b.q[1] = *(const uint4*)(pB + 16);
                acc[g] = wmma_bf16(a, b, acc[g]);
            }
        }

        wait_async0();      // xg slab landed in LDS
        __syncthreads();    // all reads of h(t-1) done; xg slab visible to all

        // ---- gates + state update ----
#pragma unroll
        for (int r8 = 0; r8 < 8; ++r8) {
            int lrow = mT * 16 + half * 8 + r8;
            const unsigned short* xr = &xgS[lrow * G4H + nT * 16 + r];
            float iv = sigf(acc[0][r8] + bf2f(xr[0 * HH]));
            float fv = sigf(acc[1][r8] + bf2f(xr[1 * HH]));
            float gv = tanhf(acc[2][r8] + bf2f(xr[2 * HH]));
            float ov = sigf(acc[3][r8] + bf2f(xr[3 * HH]));
            float cv = fv * cReg[r8] + iv * gv;
            float hv = ov * tanhf(cv);
            cReg[r8] = cv;

            int len = lenS[lrow];
            int tw = dir ? ((t < len) ? (len - 1 - t) : t) : t;
            hP[lrow * HLD + hcol] = f2bf(hv);
            size_t orow = ((size_t)tw * BB + bBase + lrow) * H2 + dir * HH + hcol;
            outBf[orow] = f2bf(hv);
            if (outF32) outF32[orow] = hv;
        }
        __syncthreads();    // h(t) + xgS reads complete before next step
    }
}

// ---------------------------------------------------------------------------
// Per-batch masked softmax over time + weighted pooling.  grid=B, block=T.
// ---------------------------------------------------------------------------
__global__ __launch_bounds__(256) void attn_pool(
    const float* __restrict__ scores,   // flat row = t*B + b
    const float* __restrict__ h2f,      // [T*B, 512]
    const int* __restrict__ lengths,
    float* __restrict__ out) {          // [B, 512]
    const int b = blockIdx.x;
    const int t = threadIdx.x;
    __shared__ float red[256];
    __shared__ float wS[256];

    int len = lengths[b];
    if (len < 1) len = 1;
    float s = scores[(size_t)t * BB + b];
    if (t >= len) s = -1e9f;

    red[t] = s;
    __syncthreads();
    for (int off = 128; off > 0; off >>= 1) {
        if (t < off) red[t] = fmaxf(red[t], red[t + off]);
        __syncthreads();
    }
    float mx = red[0];
    __syncthreads();

    float e = __expf(s - mx);
    red[t] = e;
    __syncthreads();
    for (int off = 128; off > 0; off >>= 1) {
        if (t < off) red[t] += red[t + off];
        __syncthreads();
    }
    float inv = 1.0f / red[0];
    wS[t] = e * inv;
    __syncthreads();

    for (int d = t; d < H2; d += 256) {
        float acc = 0.0f;
        for (int tt = 0; tt < TT; ++tt)
            acc += wS[tt] * h2f[((size_t)tt * BB + b) * H2 + d];
        out[(size_t)b * H2 + d] = acc;
    }
}

// ---------------------------------------------------------------------------
// Host orchestration
// ---------------------------------------------------------------------------
extern "C" void kernel_launch(void* const* d_in, const int* in_sizes, int n_in,
                              void* d_out, int out_size, void* d_ws,
                              size_t ws_size, hipStream_t stream) {
    (void)in_sizes; (void)n_in; (void)out_size; (void)ws_size;

    const float* x = (const float*)d_in[0];
    const int* lengths = (const int*)d_in[1];
    const float* w_ih_l0 = (const float*)d_in[2];
    const float* w_hh_l0 = (const float*)d_in[3];
    const float* b_ih_l0 = (const float*)d_in[4];
    const float* b_hh_l0 = (const float*)d_in[5];
    const float* w_ih_l0r = (const float*)d_in[6];
    const float* w_hh_l0r = (const float*)d_in[7];
    const float* b_ih_l0r = (const float*)d_in[8];
    const float* b_hh_l0r = (const float*)d_in[9];
    const float* w_ih_l1 = (const float*)d_in[10];
    const float* w_hh_l1 = (const float*)d_in[11];
    const float* b_ih_l1 = (const float*)d_in[12];
    const float* b_hh_l1 = (const float*)d_in[13];
    const float* w_ih_l1r = (const float*)d_in[14];
    const float* w_hh_l1r = (const float*)d_in[15];
    const float* b_ih_l1r = (const float*)d_in[16];
    const float* b_hh_l1r = (const float*)d_in[17];
    const float* mlp_w = (const float*)d_in[18];
    const float* mlp_b = (const float*)d_in[19];
    const float* ctx = (const float*)d_in[20];
    float* out = (float*)d_out;

    uintptr_t base = (uintptr_t)d_ws;
    size_t off = 0;
    auto alloc = [&](size_t bytes) -> void* {
        off = (off + 255) & ~(size_t)255;
        void* p = (void*)(base + off);
        off += bytes;
        return p;
    };

    // bf16 buffers
    unsigned short* x_bf   = (unsigned short*)alloc((size_t)MROWS * KP0 * 2);
    unsigned short* w0f_bf = (unsigned short*)alloc((size_t)G4H * KP0 * 2);
    unsigned short* w0r_bf = (unsigned short*)alloc((size_t)G4H * KP0 * 2);
    unsigned short* whh0f  = (unsigned short*)alloc((size_t)G4H * HH * 2);
    unsigned short* whh0r  = (unsigned short*)alloc((size_t)G4H * HH * 2);
    unsigned short* whh1f  = (unsigned short*)alloc((size_t)G4H * HH * 2);
    unsigned short* whh1r  = (unsigned short*)alloc((size_t)G4H * HH * 2);
    unsigned short* w1f_bf = (unsigned short*)alloc((size_t)G4H * H2 * 2);
    unsigned short* w1r_bf = (unsigned short*)alloc((size_t)G4H * H2 * 2);
    unsigned short* mlp_bf = (unsigned short*)alloc((size_t)H2 * H2 * 2);
    unsigned short* h1_bf  = (unsigned short*)alloc((size_t)MROWS * H2 * 2);
    unsigned short* h2_bf  = (unsigned short*)alloc((size_t)MROWS * H2 * 2);
    unsigned short* xgF    = (unsigned short*)alloc((size_t)MROWS * G4H * 2);
    unsigned short* xgR    = (unsigned short*)alloc((size_t)MROWS * G4H * 2);
    // fp32 buffers
    float* h2_f  = (float*)alloc((size_t)MROWS * H2 * 4);
    float* score = (float*)alloc((size_t)MROWS * 4);

    auto cvt = [&](const float* s, unsigned short* d, int K, int Kp, long N) {
        long total = N * Kp;
        int grid = (int)((total + 255) / 256);
        hipLaunchKernelGGL(cvt_pad_bf16, dim3(grid), dim3(256), 0, stream,
                           s, d, K, Kp, total);
    };
    cvt(x, x_bf, DD, KP0, MROWS);
    cvt(w_ih_l0, w0f_bf, DD, KP0, G4H);
    cvt(w_ih_l0r, w0r_bf, DD, KP0, G4H);
    cvt(w_hh_l0, whh0f, HH, HH, G4H);
    cvt(w_hh_l0r, whh0r, HH, HH, G4H);
    cvt(w_hh_l1, whh1f, HH, HH, G4H);
    cvt(w_hh_l1r, whh1r, HH, HH, G4H);
    cvt(w_ih_l1, w1f_bf, H2, H2, G4H);
    cvt(w_ih_l1r, w1r_bf, H2, H2, G4H);
    cvt(mlp_w, mlp_bf, H2, H2, H2);

    hipLaunchKernelGGL(zerof, dim3((MROWS + 255) / 256), dim3(256), 0, stream,
                       score, MROWS);

    // Layer 0 input projections (bf16 out, bias folded)
    dim3 g0(MROWS / 32, G4H / 256);
    hipLaunchKernelGGL((gemm_bf16<2>), g0, dim3(256), 0, stream,
                       x_bf, KP0, w0f_bf, KP0, b_ih_l0, b_hh_l0,
                       (float*)nullptr, xgF, G4H, KP0,
                       (const float*)nullptr, (float*)nullptr);
    hipLaunchKernelGGL((gemm_bf16<2>), g0, dim3(256), 0, stream,
                       x_bf, KP0, w0r_bf, KP0, b_ih_l0r, b_hh_l0r,
                       (float*)nullptr, xgR, G4H, KP0,
                       (const float*)nullptr, (float*)nullptr);

    // Layer 0 recurrence -> h1 (bf16)
    hipLaunchKernelGGL(lstm_bidir, dim3(8), dim3(1024), 0, stream,
                       whh0f, whh0r, xgF, xgR, lengths, h1_bf,
                       (float*)nullptr);

    // Layer 1 input projections (K = 512)
    hipLaunchKernelGGL((gemm_bf16<2>), g0, dim3(256), 0, stream,
                       h1_bf, H2, w1f_bf, H2, b_ih_l1, b_hh_l1,
                       (float*)nullptr, xgF, G4H, H2,
                       (const float*)nullptr, (float*)nullptr);
    hipLaunchKernelGGL((gemm_bf16<2>), g0, dim3(256), 0, stream,
                       h1_bf, H2, w1r_bf, H2, b_ih_l1r, b_hh_l1r,
                       (float*)nullptr, xgR, G4H, H2,
                       (const float*)nullptr, (float*)nullptr);

    // Layer 1 recurrence -> h2 (bf16 + fp32)
    hipLaunchKernelGGL(lstm_bidir, dim3(8), dim3(1024), 0, stream,
                       whh1f, whh1r, xgF, xgR, lengths, h2_bf, h2_f);

    // Attention MLP + fused ctx-dot -> scores
    dim3 g1(MROWS / 32, H2 / 256);
    hipLaunchKernelGGL((gemm_bf16<1>), g1, dim3(256), 0, stream,
                       h2_bf, H2, mlp_bf, H2, mlp_b, (const float*)nullptr,
                       (float*)nullptr, (unsigned short*)nullptr, 0, H2,
                       ctx, score);

    // Masked softmax + pooling
    hipLaunchKernelGGL(attn_pool, dim3(BB), dim3(256), 0, stream,
                       score, h2_f, lengths, out);
}
